// LSTMCellExp_50148038148717
// MI455X (gfx1250) — matrix operand order repaired and checked by reading, not compiled, not run on hardware
//
#include <hip/hip_runtime.h>

#define H 1024
#define K_TOTAL 2048
#define BM 128
#define BN 64
#define BK 64
#define LDK 72   // padded K stride (bf16 elems): 144B = 36 banks -> conflict-free b128 reads

typedef __attribute__((ext_vector_type(16))) __bf16 bf16x16;
typedef __attribute__((ext_vector_type(8)))  __bf16 bf16x8;
typedef __attribute__((ext_vector_type(8)))  float  f32x8;

__device__ __forceinline__ float sigmoid_f(float x) {
    return 1.0f / (1.0f + __expf(-x));
}
__device__ __forceinline__ float tanh_f(float x) {
    // 1 - 2/(e^{2x}+1): saturates correctly at +/-1 for large |x|
    float e = __expf(2.0f * x);
    return 1.0f - 2.0f / (e + 1.0f);
}

// Register-resident prefetch of one 64-wide K chunk (fp32)
struct Pref {
    float4 a[8];        // A tile: 128x64, this thread's 8 float4
    float4 w[4][4];     // 4 gates x (2 k-pair groups x {k, k+1}) float4 along N
};

__device__ __forceinline__ void prefetch_chunk(
    int k0, int rowBase, int colBase,
    const float* __restrict__ x, const float* __restrict__ hidden,
    const float* const* wptr, int rr, int cc, Pref& p)
{
    const float* aSrc; int aStride;
    if (k0 < H) { aSrc = x      + (size_t)rowBase * 1024 + k0;       aStride = 1024; }
    else        { aSrc = hidden + (size_t)rowBase * 2048 + (k0 - H); aStride = 2048; }
    #pragma unroll
    for (int it = 0; it < 8; ++it)
        p.a[it] = *reinterpret_cast<const float4*>(aSrc + (size_t)(rr + it * 16) * aStride + cc);
    #pragma unroll
    for (int g = 0; g < 4; ++g) {
        const float* wSrc = wptr[g] + (size_t)k0 * 1024 + colBase;
        #pragma unroll
        for (int pi = 0; pi < 2; ++pi) {
            const int k = (rr + pi * 16) * 2;                       // even k, 0..62
            p.w[g][pi * 2 + 0] = *reinterpret_cast<const float4*>(wSrc + (size_t)k       * 1024 + cc);
            p.w[g][pi * 2 + 1] = *reinterpret_cast<const float4*>(wSrc + (size_t)(k + 1) * 1024 + cc);
        }
    }
}

__device__ __forceinline__ void stage_lds(
    const Pref& p, __bf16 (*sA)[LDK], __bf16 (*sW)[BN][LDK], int rr, int cc)
{
    #pragma unroll
    for (int it = 0; it < 8; ++it) {
        union { __bf16 b[4]; uint2 u; } pk;
        pk.b[0] = (__bf16)p.a[it].x; pk.b[1] = (__bf16)p.a[it].y;
        pk.b[2] = (__bf16)p.a[it].z; pk.b[3] = (__bf16)p.a[it].w;
        *reinterpret_cast<uint2*>(&sA[rr + it * 16][cc]) = pk.u;    // ds_store_b64
    }
    #pragma unroll
    for (int g = 0; g < 4; ++g) {
        #pragma unroll
        for (int pi = 0; pi < 2; ++pi) {
            const int k = (rr + pi * 16) * 2;
            const float4 v0 = p.w[g][pi * 2 + 0];
            const float4 v1 = p.w[g][pi * 2 + 1];
            const float a0[4] = { v0.x, v0.y, v0.z, v0.w };
            const float a1[4] = { v1.x, v1.y, v1.z, v1.w };
            #pragma unroll
            for (int j = 0; j < 4; ++j) {
                union { __bf16 b[2]; unsigned u; } pk;
                pk.b[0] = (__bf16)a0[j];                            // K = k
                pk.b[1] = (__bf16)a1[j];                            // K = k+1
                *reinterpret_cast<unsigned*>(&sW[g][cc + j][k]) = pk.u;  // ds_store_b32 (transpose)
            }
        }
    }
}

__global__ __launch_bounds__(256)
void lstm_cell_wmma_bf16(const float* __restrict__ x,
                         const float* __restrict__ hidden,
                         const float* __restrict__ wi,
                         const float* __restrict__ wf,
                         const float* __restrict__ wc,
                         const float* __restrict__ wo,
                         const float* __restrict__ bi,
                         const float* __restrict__ bfv,
                         const float* __restrict__ bc,
                         const float* __restrict__ bo,
                         float* __restrict__ out)
{
    __shared__ __bf16 sA[BM][LDK];        // A tile, row-major [M][K]
    __shared__ __bf16 sW[4][BN][LDK];     // W tiles, transposed [N][K]

    const int tid   = threadIdx.x;
    const int lane  = tid & 31;
    const int wav   = tid >> 5;
    const int waveM = wav & 1;            // 2 waves along M (64 rows each)
    const int waveN = wav >> 1;           // 4 waves along N (16 cols each)
    const int m0 = waveM * 64;
    const int n0 = waveN * 16;

    const int rowBase = blockIdx.y * BM;
    const int colBase = blockIdx.x * BN;

    const float* wptr[4] = { wi, wf, wc, wo };

    f32x8 acc[4][4];                      // [gate][mi] : 64M x 16N per wave
    #pragma unroll
    for (int g = 0; g < 4; ++g)
        #pragma unroll
        for (int mi = 0; mi < 4; ++mi)
            acc[g][mi] = {};

    const int rr = tid >> 4;              // 0..15
    const int cc = (tid & 15) << 2;       // 0..60 step 4
    const int hf = lane >> 4;             // lane half (0/1)
    const int lr = lane & 15;

    Pref pref;
    prefetch_chunk(0, rowBase, colBase, x, hidden, wptr, rr, cc, pref);

    for (int k0 = 0; k0 < K_TOTAL; k0 += BK) {
        __syncthreads();                  // previous compute done before LDS overwrite
        stage_lds(pref, sA, sW, rr, cc);
        __syncthreads();

        // ---- prefetch next chunk: VMEM latency overlaps the WMMAs below ----
        if (k0 + BK < K_TOTAL)
            prefetch_chunk(k0 + BK, rowBase, colBase, x, hidden, wptr, rr, cc, pref);

        // ---- compute: 2 K-steps of 32, 16 WMMAs each ----
        #pragma unroll
        for (int ks = 0; ks < BK; ks += 32) {
            bf16x16 afrag[4];
            #pragma unroll
            for (int mi = 0; mi < 4; ++mi) {
                // A 16x32 bf16 layout: lane half selects K base (0/8), +16 for VGPR4-7
                const __bf16* pa = &sA[m0 + mi * 16 + lr][ks + hf * 8];
                bf16x8 lo = *reinterpret_cast<const bf16x8*>(pa);        // ds_load_b128
                bf16x8 hi = *reinterpret_cast<const bf16x8*>(pa + 16);   // ds_load_b128
                #pragma unroll
                for (int e = 0; e < 8; ++e) { afrag[mi][e] = lo[e]; afrag[mi][e + 8] = hi[e]; }
            }
            #pragma unroll
            for (int g = 0; g < 4; ++g) {
                // B 32x16 bf16 layout: lane half selects K base (0/8), +16 for VGPR4-7
                const __bf16* pb = &sW[g][n0 + lr][ks + hf * 8];
                bf16x8 lo = *reinterpret_cast<const bf16x8*>(pb);
                bf16x8 hi = *reinterpret_cast<const bf16x8*>(pb + 16);
                bf16x16 bfrag;
                #pragma unroll
                for (int e = 0; e < 8; ++e) { bfrag[e] = lo[e]; bfrag[e + 8] = hi[e]; }
                #pragma unroll
                for (int mi = 0; mi < 4; ++mi)
                    acc[g][mi] = __builtin_amdgcn_wmma_f32_16x16x32_bf16(
                        false, afrag[mi], false, bfrag, (short)0,
                        acc[g][mi], false, false);
            }
        }
    }

    // ---- fused LSTM epilogue ----
    const float* bptr[4] = { bi, bfv, bc, bo };
    const int col = colBase + n0 + lr;                  // N = lane%16
    float bs[4];
    #pragma unroll
    for (int g = 0; g < 4; ++g) bs[g] = bptr[g][col] + bptr[g][H + col];

    #pragma unroll
    for (int mi = 0; mi < 4; ++mi) {
        const int rbase = rowBase + m0 + mi * 16 + hf * 8;   // M = r (+8 for upper lane half)
        #pragma unroll
        for (int r = 0; r < 8; ++r) {
            const size_t row = (size_t)(rbase + r);
            const float zi = acc[0][mi][r] + bs[0];
            const float zf = acc[1][mi][r] + bs[1];
            const float zg = acc[2][mi][r] + bs[2];
            const float zo = acc[3][mi][r] + bs[3];
            const float cprev = hidden[row * 2048 + H + col];
            const float it = sigmoid_f(zi);
            const float ft = sigmoid_f(zf);
            const float gt = tanh_f(zg);
            const float ot = sigmoid_f(zo);
            const float c  = ft * cprev + it * gt;
            const float h  = ot * tanh_f(c);
            out[row * 2048 + col]     = h;
            out[row * 2048 + H + col] = c;
        }
    }
}

extern "C" void kernel_launch(void* const* d_in, const int* in_sizes, int n_in,
                              void* d_out, int out_size, void* d_ws, size_t ws_size,
                              hipStream_t stream) {
    // setup_inputs order: x, hidden_state, weight_i, weight_f, weight_o, weight,
    //                     bias_i, bias_f, bias_o, bias
    const float* x      = (const float*)d_in[0];
    const float* hidden = (const float*)d_in[1];
    const float* wi     = (const float*)d_in[2];
    const float* wf     = (const float*)d_in[3];
    const float* wo     = (const float*)d_in[4];
    const float* wc     = (const float*)d_in[5];
    const float* bi     = (const float*)d_in[6];
    const float* bf_    = (const float*)d_in[7];
    const float* bo     = (const float*)d_in[8];
    const float* bc     = (const float*)d_in[9];
    float* out = (float*)d_out;

    dim3 grid(1024 / BN, 8192 / BM);   // (16, 64)
    dim3 block(256);
    hipLaunchKernelGGL(lstm_cell_wmma_bf16, grid, block, 0, stream,
                       x, hidden, wi, wf, wc, wo, bi, bf_, bc, bo, out);
}
